// SlidingWindowAttention_40827959116498
// MI455X (gfx1250) — matrix operand order repaired and checked by reading, not compiled, not run on hardware
//
#include <hip/hip_runtime.h>

// ---------------------------------------------------------------------------
// Sliding-window causal attention (D=1024, H=16, Dh=64, WINDOW=256, B=2, S=2048)
// bf16 WMMA (v_wmma_f32_16x16x32_bf16) everywhere, f32 accumulation.
// GEMM tiles staged by the Tensor Data Mover (tensor_load_to_lds), double
// buffered with s_wait_tensorcnt.  Q/K/V/Ctx kept in bf16 in workspace.
// ---------------------------------------------------------------------------

typedef __attribute__((ext_vector_type(16))) __bf16        v16bf;
typedef __attribute__((ext_vector_type(8)))  float         v8f;
typedef __attribute__((ext_vector_type(4)))  unsigned int  v4u;
typedef __attribute__((ext_vector_type(8)))  int           v8i;
typedef __attribute__((ext_vector_type(4)))  int           v4i;

union BF16Frag { uint4 q[2]; v16bf v; };

__device__ __forceinline__ unsigned f2bf2(float lo, float hi) {
  union { float f; unsigned u; } a, b;
  a.f = lo; b.f = hi;
  unsigned ul = a.u + 0x7FFFu + ((a.u >> 16) & 1u);   // RNE
  unsigned uh = b.u + 0x7FFFu + ((b.u >> 16) & 1u);
  return (ul >> 16) | (uh & 0xFFFF0000u);
}

__device__ __forceinline__ unsigned short f2bf1(float f) {
  union { float fl; unsigned u; } a; a.fl = f;
  return (unsigned short)((a.u + 0x7FFFu + ((a.u >> 16) & 1u)) >> 16);
}

__device__ __forceinline__ uint4 pack8(float4 a, float4 b) {
  uint4 r;
  r.x = f2bf2(a.x, a.y); r.y = f2bf2(a.z, a.w);
  r.z = f2bf2(b.x, b.y); r.w = f2bf2(b.z, b.w);
  return r;
}

__device__ __forceinline__ v8f v8zero() {
  v8f v;
#pragma unroll
  for (int i = 0; i < 8; ++i) v[i] = 0.0f;
  return v;
}

__device__ __forceinline__ unsigned lds_byte_off(const void* p) {
  // Generic LDS pointer = {shared aperture, 32-bit LDS offset}; low 32 bits
  // are exactly what DS / the TDM D# lds_addr field consume.
  return (unsigned)(unsigned long long)p;
}

// Issue one 2D TDM load: tile_x x tile_y bf16 elements, row stride stride_x
// elements in global memory, optional LDS row padding (ISA 8.3-8.5 D# layout).
__device__ __forceinline__ void tdm_load_2d(unsigned lds_addr, const void* gptr,
                                            unsigned tile_x, unsigned tile_y,
                                            unsigned stride_x,
                                            unsigned pad_int_code,
                                            unsigned pad_amt_code,
                                            bool pad_en) {
  const unsigned long long ga = (unsigned long long)gptr;
  v4u g0;
  g0[0] = 1u;                                             // count=1, user mode
  g0[1] = lds_addr;                                       // lds_addr [63:32]
  g0[2] = (unsigned)ga;                                   // global_addr lo
  g0[3] = (unsigned)((ga >> 32) & 0x1FFFFFFu) | (2u << 30);  // addr hi | type=2
  v8i g1;
  g1[0] = (int)((1u << 16) |                              // data_size = 2B
                (pad_en ? (1u << 20) : 0u) |
                (pad_int_code << 22) | (pad_amt_code << 25));
  g1[1] = (int)((tile_x & 0xFFFFu) << 16);                // tensor_dim0 lo
  g1[2] = (int)((tile_x >> 16) | ((tile_y & 0xFFFFu) << 16)); // td0 hi|td1 lo
  g1[3] = (int)((tile_y >> 16) | (tile_x << 16));         // td1 hi | tile_dim0
  g1[4] = (int)(tile_y & 0xFFFFu);                        // tile_dim1, dim2=0
  g1[5] = (int)stride_x;                                  // tensor_dim0_stride
  g1[6] = 0;
  g1[7] = 0;
  const v4i z4 = {0, 0, 0, 0};
  const v8i z8 = {0, 0, 0, 0, 0, 0, 0, 0};
  __builtin_amdgcn_tensor_load_to_lds(g0, g1, z4, z4, z8, 0);
}

// ---------------------------------------------------------------------------
// Kernel 0: f32 -> bf16 conversion (grid stride, 8 elems/thread/iter)
// ---------------------------------------------------------------------------
__global__ void cvt_bf16(const float* __restrict__ in,
                         unsigned short* __restrict__ outp, long long n) {
  long long i = (blockIdx.x * (long long)blockDim.x + threadIdx.x) * 8;
  const long long stride = gridDim.x * (long long)blockDim.x * 8;
  for (; i < n; i += stride) {
    const float4* g = (const float4*)(in + i);
    *(uint4*)(outp + i) = pack8(g[0], g[1]);
  }
}

// ---------------------------------------------------------------------------
// Kernel 1: out[M,N] = A[M,K](bf16) * W[N,K](bf16)^T + bias[N](f32)
// 64x64 tile/block, 128 threads = 4 waves.  TDM-staged LDS tiles, double
// buffered; wave 0 drives the Tensor Data Mover.
// ---------------------------------------------------------------------------
#define G_BM 64
#define G_BN 64
#define G_BK 32
#define G_LDT 40   // bf16/row in LDS (32 + 4dword TDM pad); 80B row (16B align)

template <bool OUT_BF16>
__global__ __launch_bounds__(128)
void gemm_bias_tdm(const unsigned short* __restrict__ A,
                   const unsigned short* __restrict__ W,
                   const float* __restrict__ bias, void* __restrict__ outp,
                   int M, int N, int K) {
  __shared__ unsigned short As[2][G_BM * G_LDT];
  __shared__ unsigned short Bs[2][G_BN * G_LDT];

  const int t    = threadIdx.x;
  const int wv   = t >> 5;
  const int lane = t & 31;
  const int half = lane >> 4;
  const int l15  = lane & 15;
  const int m0   = blockIdx.x * G_BM;
  const int n0   = blockIdx.y * G_BN;

  v8f acc[4];
#pragma unroll
  for (int j = 0; j < 4; ++j) acc[j] = v8zero();

  const int nIter = K / G_BK;
  // prologue: fill buffer 0 (pad: 16 dwords interval -> code 3; 4 dwords pad)
  if (wv == 0) {
    tdm_load_2d(lds_byte_off(&As[0][0]), A + (size_t)m0 * K, G_BK, G_BM, K, 3, 3, true);
    tdm_load_2d(lds_byte_off(&Bs[0][0]), W + (size_t)n0 * K, G_BK, G_BM, K, 3, 3, true);
  }

  for (int i = 0; i < nIter; ++i) {
    if (wv == 0) __builtin_amdgcn_s_wait_tensorcnt(0);
    __syncthreads();                       // tiles for step i visible to all
    if (wv == 0 && i + 1 < nIter) {        // overlap: DMA step i+1
      const int k1 = (i + 1) * G_BK;
      tdm_load_2d(lds_byte_off(&As[(i + 1) & 1][0]),
                  A + (size_t)m0 * K + k1, G_BK, G_BM, K, 3, 3, true);
      tdm_load_2d(lds_byte_off(&Bs[(i + 1) & 1][0]),
                  W + (size_t)n0 * K + k1, G_BK, G_BM, K, 3, 3, true);
    }
    const int buf = i & 1;
    // A fragment: row = 16*wv + l15; K runs {8h..8h+7},{16+8h..+7}
    BF16Frag af;
    {
      const char* ab = (const char*)&As[buf][0] + (wv * 16 + l15) * (G_LDT * 2) + half * 16;
      af.q[0] = *(const uint4*)(ab);
      af.q[1] = *(const uint4*)(ab + 32);
    }
#pragma unroll
    for (int j = 0; j < 4; ++j) {
      BF16Frag bf;
      const char* bb = (const char*)&Bs[buf][0] + (j * 16 + l15) * (G_LDT * 2) + half * 16;
      bf.q[0] = *(const uint4*)(bb);
      bf.q[1] = *(const uint4*)(bb + 32);
      acc[j] = __builtin_amdgcn_wmma_f32_16x16x32_bf16(
          false, af.v, false, bf.v, (short)0, acc[j], false, false);
    }
  }
  __syncthreads();

  // epilogue: C layout row = v + 8*half, col = l15
#pragma unroll
  for (int j = 0; j < 4; ++j) {
    const int n = n0 + j * 16 + l15;
    const float bval = bias[n];
#pragma unroll
    for (int v = 0; v < 8; ++v) {
      const int m = m0 + wv * 16 + v + 8 * half;
      const float r = acc[j][v] + bval;
      if (OUT_BF16)
        ((unsigned short*)outp)[(size_t)m * N + n] = f2bf1(r);
      else
        ((float*)outp)[(size_t)m * N + n] = r;
    }
  }
}

// ---------------------------------------------------------------------------
// Kernel 2: sliding-window causal attention (bf16 Q/K/V in global).
// Block = 64 threads (2 waves), 32 queries/block, grid = (S/32, B*H).
// Pass 1: WMMA scores -> masked/scaled -> per-wave LDS.
// Stats:  per-lane half-row scans (2 shuffles/row total).
// Write:  full 2048-wide attn rows (single pass, float4, zeros off-band).
// Pass 2: P rebuilt in A layout from LDS, ctx = P*V via WMMA.
// ---------------------------------------------------------------------------
#define AT_Q   32
#define AT_LDS 292   // f32 per score row (288+4); 1168B stride (16B-aligned)
#define AT_LDV 40    // bf16 per V^T LDS row (32+8); 80B stride (16B-aligned)

__global__ __launch_bounds__(64)
void swattn_bf16(const unsigned short* __restrict__ Qb,
                 const unsigned short* __restrict__ Kb,
                 const unsigned short* __restrict__ Vb,
                 float* __restrict__ attn,
                 unsigned short* __restrict__ Ctx) {
  __shared__ unsigned short Vs[64 * AT_LDV];   // 5120 B (V^T, one 32-key block)
  __shared__ float Sw[2][16 * AT_LDS];         // 37376 B (per-wave scores)
  __shared__ float stats[2][16][2];            // m, 1/sum per row

  const int t    = threadIdx.x;
  const int wv   = t >> 5;
  const int lane = t & 31;
  const int half = lane >> 4;
  const int l15  = lane & 15;

  const int b = blockIdx.y >> 4;
  const int h = blockIdx.y & 15;
  const int q0 = blockIdx.x * AT_Q;
  const int k_lo = (q0 >= 256) ? (q0 - 256) : 0;
  const int nk = q0 + AT_Q - k_lo;             // multiple of 32, <= 288

  const size_t headoff = (size_t)b * 2048 * 1024 + (size_t)h * 64;
  const float scale = 0.125f;                  // 64^-0.5

  // ---- Q A-fragments (two 32-dim blocks), direct from global bf16 ----
  BF16Frag qa[2];
  {
    const unsigned short* qrow = Qb + headoff + (size_t)(q0 + wv * 16 + l15) * 1024;
#pragma unroll
    for (int d = 0; d < 2; ++d) {
      qa[d].q[0] = *(const uint4*)(qrow + d * 32 + 8 * half);
      qa[d].q[1] = *(const uint4*)(qrow + d * 32 + 16 + 8 * half);
    }
  }

  // ================= pass 1: scores -> masked -> LDS =================
  for (int kc = k_lo; kc < k_lo + nk; kc += 16) {
    const unsigned short* krow = Kb + headoff + (size_t)(kc + l15) * 1024;
    v8f sc = v8zero();
#pragma unroll
    for (int d = 0; d < 2; ++d) {    // B = K^T: lane col = key, K dim = dims
      BF16Frag kf;
      kf.q[0] = *(const uint4*)(krow + d * 32 + 8 * half);
      kf.q[1] = *(const uint4*)(krow + d * 32 + 16 + 8 * half);
      sc = __builtin_amdgcn_wmma_f32_16x16x32_bf16(
          false, qa[d].v, false, kf.v, (short)0, sc, false, false);
    }
    const int scol = kc - k_lo;
#pragma unroll
    for (int v = 0; v < 8; ++v) {
      const int qi = q0 + wv * 16 + v + 8 * half;
      const int kj = kc + l15;
      const bool ok = (kj <= qi) && (qi - kj < 256);
      Sw[wv][(v + 8 * half) * AT_LDS + scol + l15] = ok ? sc[v] * scale : -3.0e38f;
    }
  }

  // ====== stats: lane = (row l15, half-row seg); 2 shuffles per row ======
  {
    const int seg = nk >> 1;                        // multiple of 16
    const float* srow = &Sw[wv][l15 * AT_LDS + half * seg];
    float m = -1.0e30f;
    for (int c = 0; c < seg; c += 4) {
      const float4 s = *(const float4*)(srow + c);
      m = fmaxf(m, fmaxf(fmaxf(s.x, s.y), fmaxf(s.z, s.w)));
    }
    m = fmaxf(m, __shfl_xor(m, 16, 32));            // combine half-rows
    float sum = 0.0f;
    for (int c = 0; c < seg; c += 4) {
      const float4 s = *(const float4*)(srow + c);
      sum += __expf(s.x - m) + __expf(s.y - m) + __expf(s.z - m) + __expf(s.w - m);
    }
    sum += __shfl_xor(sum, 16, 32);
    if (half == 0) {
      stats[wv][l15][0] = m;
      stats[wv][l15][1] = 1.0f / sum;
    }
  }
  __syncthreads();

  // ====== write attn: full 2048-wide rows, single pass, float4 ======
  {
    float* abase = attn + ((size_t)(b * 16 + h) * 2048) * 2048;
#pragma unroll 1
    for (int r = 0; r < 16; ++r) {
      const int qi = q0 + wv * 16 + r;
      const float mr = stats[wv][r][0];
      const float ir = stats[wv][r][1];
      float* orow = abase + (size_t)qi * 2048;
      const float* srow = &Sw[wv][r * AT_LDS];
#pragma unroll 1
      for (int c4 = lane * 4; c4 < 2048; c4 += 128) {
        const int cr = c4 - k_lo;        // k_lo, nk multiples of 32 -> group
        float4 o;                        // is entirely in-band or off-band
        if (cr >= 0 && cr < nk) {
          const float4 s = *(const float4*)(srow + cr);
          o.x = __expf(s.x - mr) * ir;
          o.y = __expf(s.y - mr) * ir;
          o.z = __expf(s.z - mr) * ir;
          o.w = __expf(s.w - mr) * ir;
        } else {
          o.x = 0.f; o.y = 0.f; o.z = 0.f; o.w = 0.f;
        }
        *(float4*)(orow + c4) = o;
      }
    }
  }

  // ================= pass 2: ctx = P * V via WMMA =================
  v8f ctx[4];
#pragma unroll
  for (int j = 0; j < 4; ++j) ctx[j] = v8zero();

  const float mr = stats[wv][l15][0];   // A-layout: lane owns row l15
  const float ir = stats[wv][l15][1];

  for (int kb = k_lo; kb < k_lo + nk; kb += 32) {
    __syncthreads();
    {   // stage V^T (64 dims x 32 keys) from bf16 global
      const int key = t >> 1, db = (t & 1) * 32;
      const uint4* g = (const uint4*)(Vb + headoff + (size_t)(kb + key) * 1024 + db);
      uint4 r0 = g[0], r1 = g[1], r2 = g[2], r3 = g[3];
      const unsigned u[16] = {r0.x, r0.y, r0.z, r0.w, r1.x, r1.y, r1.z, r1.w,
                              r2.x, r2.y, r2.z, r2.w, r3.x, r3.y, r3.z, r3.w};
#pragma unroll
      for (int i = 0; i < 16; ++i) {
        const int d = db + 2 * i;
        Vs[(d + 0) * AT_LDV + key] = (unsigned short)u[i];
        Vs[(d + 1) * AT_LDV + key] = (unsigned short)(u[i] >> 16);
      }
    }
    __syncthreads();

    // rebuild normalized P in A layout from stored scores
    BF16Frag pf;
    {
      const float* srow = &Sw[wv][l15 * AT_LDS + (kb - k_lo)];
      float4 s0 = *(const float4*)(srow + 8 * half);
      float4 s1 = *(const float4*)(srow + 8 * half + 4);
      float4 s2 = *(const float4*)(srow + 16 + 8 * half);
      float4 s3 = *(const float4*)(srow + 16 + 8 * half + 4);
      float p[16];
      p[0]  = __expf(s0.x - mr) * ir;  p[1]  = __expf(s0.y - mr) * ir;
      p[2]  = __expf(s0.z - mr) * ir;  p[3]  = __expf(s0.w - mr) * ir;
      p[4]  = __expf(s1.x - mr) * ir;  p[5]  = __expf(s1.y - mr) * ir;
      p[6]  = __expf(s1.z - mr) * ir;  p[7]  = __expf(s1.w - mr) * ir;
      p[8]  = __expf(s2.x - mr) * ir;  p[9]  = __expf(s2.y - mr) * ir;
      p[10] = __expf(s2.z - mr) * ir;  p[11] = __expf(s2.w - mr) * ir;
      p[12] = __expf(s3.x - mr) * ir;  p[13] = __expf(s3.y - mr) * ir;
      p[14] = __expf(s3.z - mr) * ir;  p[15] = __expf(s3.w - mr) * ir;
      union { unsigned u[8]; uint4 q[2]; } pk;
#pragma unroll
      for (int i = 0; i < 8; ++i) pk.u[i] = f2bf2(p[2 * i], p[2 * i + 1]);
      pf.q[0] = pk.q[0]; pf.q[1] = pk.q[1];
    }

#pragma unroll
    for (int j = 0; j < 4; ++j) {   // B = V block: lane col = dim, K = keys
      BF16Frag vb;
      const char* bb = (const char*)Vs + (j * 16 + l15) * (AT_LDV * 2) + half * 16;
      vb.q[0] = *(const uint4*)(bb);
      vb.q[1] = *(const uint4*)(bb + 32);
      ctx[j] = __builtin_amdgcn_wmma_f32_16x16x32_bf16(
          false, pf.v, false, vb.v, (short)0, ctx[j], false, false);
    }
  }

  // ---- write context bf16 (B,S,1024), n = h*64 + d ----
#pragma unroll
  for (int j = 0; j < 4; ++j) {
#pragma unroll
    for (int v = 0; v < 8; ++v) {
      const int row = q0 + wv * 16 + v + 8 * half;
      Ctx[headoff + (size_t)row * 1024 + j * 16 + l15] = f2bf1(ctx[j][v]);
    }
  }
}

// ---------------------------------------------------------------------------
// Launch.  ws (bf16/ushort): xb(8MB) Wq/Wk/Wv/Wo(2MB ea) Q/K/V(8MB ea) Ctx(8MB)
// d_out: [out: 2*2048*1024 f32][attn: 2*16*2048*2048 f32]
// ---------------------------------------------------------------------------
extern "C" void kernel_launch(void* const* d_in, const int* in_sizes, int n_in,
                              void* d_out, int out_size, void* d_ws, size_t ws_size,
                              hipStream_t stream) {
  (void)in_sizes; (void)n_in; (void)out_size; (void)ws_size;
  const float* x  = (const float*)d_in[0];
  const float* Wq = (const float*)d_in[1];
  const float* bq = (const float*)d_in[2];
  const float* Wk = (const float*)d_in[3];
  const float* bk = (const float*)d_in[4];
  const float* Wv = (const float*)d_in[5];
  const float* bv = (const float*)d_in[6];
  const float* Wo = (const float*)d_in[7];
  const float* bo = (const float*)d_in[8];

  float* out  = (float*)d_out;
  float* attn = out + (size_t)2 * 2048 * 1024;

  unsigned short* xb  = (unsigned short*)d_ws;
  unsigned short* wqb = xb  + (size_t)4096 * 1024;
  unsigned short* wkb = wqb + (size_t)1024 * 1024;
  unsigned short* wvb = wkb + (size_t)1024 * 1024;
  unsigned short* wob = wvb + (size_t)1024 * 1024;
  unsigned short* Qd  = wob + (size_t)1024 * 1024;
  unsigned short* Kd  = Qd  + (size_t)4096 * 1024;
  unsigned short* Vd  = Kd  + (size_t)4096 * 1024;
  unsigned short* Cd  = Vd  + (size_t)4096 * 1024;

  cvt_bf16<<<1024, 256, 0, stream>>>(x,  xb,  (long long)4096 * 1024);
  cvt_bf16<<<512,  256, 0, stream>>>(Wq, wqb, (long long)1024 * 1024);
  cvt_bf16<<<512,  256, 0, stream>>>(Wk, wkb, (long long)1024 * 1024);
  cvt_bf16<<<512,  256, 0, stream>>>(Wv, wvb, (long long)1024 * 1024);
  cvt_bf16<<<512,  256, 0, stream>>>(Wo, wob, (long long)1024 * 1024);

  const dim3 gG(4096 / G_BM, 1024 / G_BN), tG(128);
  gemm_bias_tdm<true><<<gG, tG, 0, stream>>>(xb, wqb, bq, Qd, 4096, 1024, 1024);
  gemm_bias_tdm<true><<<gG, tG, 0, stream>>>(xb, wkb, bk, Kd, 4096, 1024, 1024);
  gemm_bias_tdm<true><<<gG, tG, 0, stream>>>(xb, wvb, bv, Vd, 4096, 1024, 1024);

  const dim3 gA(2048 / AT_Q, 2 * 16), tA(64);
  swattn_bf16<<<gA, tA, 0, stream>>>(Qd, Kd, Vd, attn, Cd);

  gemm_bias_tdm<false><<<gG, tG, 0, stream>>>(Cd, wob, bo, out, 4096, 1024, 1024);
}
